// AreaAttention_84112639525483
// MI455X (gfx1250) — matrix operand order repaired and checked by reading, not compiled, not run on hardware
//
#include <hip/hip_runtime.h>

typedef __attribute__((ext_vector_type(16))) _Float16 v16h;
typedef __attribute__((ext_vector_type(8)))  _Float16 v8h;
typedef __attribute__((ext_vector_type(8)))  float    v8f;

#define Bn    16
#define Cc    256
#define Hh    96
#define Ww    96
#define CQ    32
#define HW    (Hh*Ww)         /* 9216 */
#define NPIX  (Bn*HW)         /* 147456 */
#define NOUT  (CQ+CQ+Cc)      /* 320 fused output channels: q|k|v */
#define NTIL  (NOUT/16)       /* 20 */
#define KST   (Cc/32)         /* 8 K-steps of 32 */

/* workspace layout (bytes) */
#define FRAG_OFF  0                                  /* 20*8*32*16 halves = 160KB */
#define Q_OFF     (size_t)262144
#define K_OFF     (Q_OFF + (size_t)NPIX*CQ*2)        /* +9.4MB */
#define V_OFF     (K_OFF + (size_t)NPIX*CQ*2)        /* +9.4MB ; v = 75.5MB */

/* ---------------------------------------------------------------------------
 * Kernel 1: fuse Wq(32x256), Wk(32x256), Wv(256x256) into f16 B-fragments,
 * pre-swizzled to this code's B(32x16) lane layout (contiguous k, same as A) so the GEMM
 * kernel loads one contiguous 32B chunk per lane per (ntile, kstep).
 * Fragment element e (0..15) of lane L: N = L&15, K = (L>>4)*16 + e.
 * ------------------------------------------------------------------------- */
__global__ void __launch_bounds__(256) prep_weights(
    const float* __restrict__ Wq, const float* __restrict__ Wk,
    const float* __restrict__ Wv, _Float16* __restrict__ frag)
{
    int t = blockIdx.x * blockDim.x + threadIdx.x;      /* 0 .. 20*8*32-1 */
    if (t >= NTIL * KST * 32) return;
    int lane  = t & 31;
    int ks    = (t >> 5) & 7;
    int nt    = t >> 8;
    int n     = lane & 15;
    int khalf = lane >> 4;
    int o     = nt * 16 + n;                            /* fused out channel */
#pragma unroll
    for (int e = 0; e < 16; ++e) {
        int kk = ks * 32 + khalf * 16 + e;              /* channel index */
        float w;
        if (o < CQ)           w = Wq[o * Cc + kk];
        else if (o < 2 * CQ)  w = Wk[(o - CQ) * Cc + kk];
        else                  w = Wv[(o - 2 * CQ) * Cc + kk];
        frag[(size_t)t * 16 + e] = (_Float16)w;
    }
}

/* ---------------------------------------------------------------------------
 * Kernel 2: fused q/k/v projection GEMM with v_wmma_f32_16x16x32_f16.
 * Each wave: 16 pixels (M) x 320 channels (20 N-tiles), K=256 (8 steps).
 * A layout (16-bit 16x32): lanes 0-15 M=lane, K 0..15;
 * lanes 16-31 M=lane-16, K 16..31 (matches B frag).  Bias folded into accumulator.
 * Destination (q/k/v) selection is uniform in nt -> branch-free stores.
 * ------------------------------------------------------------------------- */
__global__ void __launch_bounds__(256) proj_gemm(
    const float* __restrict__ x, const _Float16* __restrict__ frag,
    const float* __restrict__ bq, const float* __restrict__ bk,
    const float* __restrict__ bv,
    _Float16* __restrict__ qb, _Float16* __restrict__ kb,
    _Float16* __restrict__ vb)
{
    const int lane  = threadIdx.x & 31;
    const int wave  = threadIdx.x >> 5;
    const int Pbase = blockIdx.x * 128 + wave * 16;     /* 128 pixels / block */
    const int m     = lane & 15;
    const int hi    = lane >> 4;
    const int P     = Pbase + m;                        /* this lane's A row  */
    const int b     = P / HW;                           /* 9216 % 128 == 0: no straddle */
    const int pix   = P - b * HW;
    const float* xb = x + (size_t)b * Cc * HW + pix;

    /* Load all A fragments for K=256 (coalesced: 16 lanes = 16 consecutive pixels) */
    v16h a[KST];
#pragma unroll
    for (int ks = 0; ks < KST; ++ks) {
#pragma unroll
        for (int j = 0; j < 8; ++j) {
            int kbase = 2 * j + hi * 16;
            int c0 = ks * 32 + kbase;
            a[ks][2 * j]     = (_Float16)xb[(size_t)c0 * HW];
            a[ks][2 * j + 1] = (_Float16)xb[(size_t)(c0 + 1) * HW];
        }
    }

    const v16h* fragv = (const v16h*)frag;
#pragma unroll 1
    for (int nt = 0; nt < NTIL; ++nt) {
        /* uniform (scalar) per-tile destination selection */
        _Float16* dst; const float* bias_p; int stride; int cb;
        if (nt < 2)      { dst = qb; bias_p = bq; stride = CQ; cb = nt * 16; }
        else if (nt < 4) { dst = kb; bias_p = bk; stride = CQ; cb = (nt - 2) * 16; }
        else             { dst = vb; bias_p = bv; stride = Cc; cb = (nt - 4) * 16; }

        float bias = bias_p[cb + m];                    /* C/D: channel N = lane&15 */
        v8f acc;
#pragma unroll
        for (int r = 0; r < 8; ++r) acc[r] = bias;
#pragma unroll
        for (int ks = 0; ks < KST; ++ks) {
            v16h bf = fragv[(nt * KST + ks) * 32 + lane];  /* contiguous 32B per lane */
            acc = __builtin_amdgcn_wmma_f32_16x16x32_f16(
                      false, a[ks], false, bf, (short)0, acc, false, false);
        }
        /* C/D layout: M = r + 8*hi ; 16 lanes store 16 consecutive channels (32B) */
#pragma unroll
        for (int r = 0; r < 8; ++r) {
            int Prow = Pbase + r + hi * 8;
            dst[(size_t)Prow * stride + cb + m] = (_Float16)acc[r];
        }
    }
}

/* ---------------------------------------------------------------------------
 * Kernel 3: per-thread 3x3 area-attention tile (self-contained: horizontal
 * band attention needs only the tile's 3 rows per column, vertical only its
 * 3 columns per row).  q/k/v read 8 channels per b128 load (pixel-major).
 * Fuses gamma*(out_h+out_v)+x into the final store.
 * ------------------------------------------------------------------------- */
__global__ void __launch_bounds__(256) area_attn(
    const _Float16* __restrict__ qb, const _Float16* __restrict__ kb,
    const _Float16* __restrict__ vb, const float* __restrict__ x,
    const float* __restrict__ gamma, float* __restrict__ out)
{
    int t = blockIdx.x * blockDim.x + threadIdx.x;
    if (t >= Bn * 32 * 32) return;
    int b  = t >> 10;
    int r  = t & 1023;
    int nh = r >> 5;
    int nw = r & 31;
    int h0 = nh * 3, w0 = nw * 3;

    int Pid[3][3];
#pragma unroll
    for (int i = 0; i < 3; ++i)
#pragma unroll
        for (int j = 0; j < 3; ++j)
            Pid[i][j] = b * HW + (h0 + i) * Ww + (w0 + j);

    float sh[3][3][3];   /* [col w][k-row][j-row] */
    float sv[3][3][3];   /* [row i][k-col][j-col] */
#pragma unroll
    for (int i = 0; i < 3; ++i)
#pragma unroll
        for (int kk = 0; kk < 3; ++kk)
#pragma unroll
            for (int j = 0; j < 3; ++j) { sh[i][kk][j] = 0.f; sv[i][kk][j] = 0.f; }

    /* scores: 8 channels per chunk, one 16B load per pixel per tensor */
#pragma unroll 1
    for (int c0 = 0; c0 < CQ; c0 += 8) {
        v8h qv8[3][3], kv8[3][3];
#pragma unroll
        for (int i = 0; i < 3; ++i)
#pragma unroll
            for (int j = 0; j < 3; ++j) {
                qv8[i][j] = *(const v8h*)(qb + (size_t)Pid[i][j] * CQ + c0);
                kv8[i][j] = *(const v8h*)(kb + (size_t)Pid[i][j] * CQ + c0);
            }
#pragma unroll
        for (int cl = 0; cl < 8; ++cl) {
            float qv[3][3], kv[3][3];
#pragma unroll
            for (int i = 0; i < 3; ++i)
#pragma unroll
                for (int j = 0; j < 3; ++j) {
                    qv[i][j] = (float)qv8[i][j][cl];
                    kv[i][j] = (float)kv8[i][j][cl];
                }
#pragma unroll
            for (int w = 0; w < 3; ++w)
#pragma unroll
                for (int kk = 0; kk < 3; ++kk)
#pragma unroll
                    for (int j = 0; j < 3; ++j) {
                        sh[w][kk][j] += qv[kk][w] * kv[j][w];   /* along rows  */
                        sv[w][kk][j] += qv[w][kk] * kv[w][j];   /* along cols  */
                    }
        }
    }

    /* stable softmax over last index j */
#pragma unroll
    for (int a0 = 0; a0 < 3; ++a0)
#pragma unroll
        for (int kk = 0; kk < 3; ++kk) {
            float mh = fmaxf(sh[a0][kk][0], fmaxf(sh[a0][kk][1], sh[a0][kk][2]));
            float e0 = __expf(sh[a0][kk][0] - mh);
            float e1 = __expf(sh[a0][kk][1] - mh);
            float e2 = __expf(sh[a0][kk][2] - mh);
            float inv = 1.f / (e0 + e1 + e2);
            sh[a0][kk][0] = e0 * inv; sh[a0][kk][1] = e1 * inv; sh[a0][kk][2] = e2 * inv;

            float mv = fmaxf(sv[a0][kk][0], fmaxf(sv[a0][kk][1], sv[a0][kk][2]));
            float f0 = __expf(sv[a0][kk][0] - mv);
            float f1 = __expf(sv[a0][kk][1] - mv);
            float f2 = __expf(sv[a0][kk][2] - mv);
            float inw = 1.f / (f0 + f1 + f2);
            sv[a0][kk][0] = f0 * inw; sv[a0][kk][1] = f1 * inw; sv[a0][kk][2] = f2 * inw;
        }

    float g = gamma[0];
#pragma unroll 1
    for (int c0 = 0; c0 < Cc; c0 += 8) {
        v8h vv8[3][3];
#pragma unroll
        for (int i = 0; i < 3; ++i)
#pragma unroll
            for (int j = 0; j < 3; ++j)
                vv8[i][j] = *(const v8h*)(vb + (size_t)Pid[i][j] * Cc + c0);
#pragma unroll
        for (int cl = 0; cl < 8; ++cl) {
            int c = c0 + cl;
            size_t cbase = ((size_t)(b * Cc + c)) * HW + (size_t)h0 * Ww + w0;
#pragma unroll
            for (int i = 0; i < 3; ++i)
#pragma unroll
                for (int j = 0; j < 3; ++j) {
                    float oh = 0.f, ov = 0.f;
#pragma unroll
                    for (int kk = 0; kk < 3; ++kk) {
                        oh += (float)vv8[kk][j][cl] * sh[j][kk][i];  /* out_h */
                        ov += (float)vv8[i][kk][cl] * sv[i][kk][j];  /* out_v */
                    }
                    size_t oidx = cbase + (size_t)i * Ww + j;
                    out[oidx] = g * (oh + ov) + x[oidx];
                }
        }
    }
}

extern "C" void kernel_launch(void* const* d_in, const int* in_sizes, int n_in,
                              void* d_out, int out_size, void* d_ws, size_t ws_size,
                              hipStream_t stream) {
    (void)in_sizes; (void)n_in; (void)out_size; (void)ws_size;
    const float* x     = (const float*)d_in[0];
    const float* Wq    = (const float*)d_in[1];
    const float* bq    = (const float*)d_in[2];
    const float* Wk    = (const float*)d_in[3];
    const float* bk    = (const float*)d_in[4];
    const float* Wv    = (const float*)d_in[5];
    const float* bv    = (const float*)d_in[6];
    const float* gamma = (const float*)d_in[7];
    float* out = (float*)d_out;

    char* ws = (char*)d_ws;
    _Float16* frag = (_Float16*)(ws + FRAG_OFF);
    _Float16* qbuf = (_Float16*)(ws + Q_OFF);
    _Float16* kbuf = (_Float16*)(ws + K_OFF);
    _Float16* vbuf = (_Float16*)(ws + V_OFF);

    /* 1: weight fusion + fragment swizzle (20*8*32 = 5120 threads) */
    prep_weights<<<20, 256, 0, stream>>>(Wq, Wk, Wv, frag);
    /* 2: WMMA projection GEMM: 147456 pixels / 128 per block */
    proj_gemm<<<NPIX / 128, 256, 0, stream>>>(x, frag, bq, bk, bv, qbuf, kbuf, vbuf);
    /* 3: fused 3x3 dual-direction attention + residual: 16384 tiles */
    area_attn<<<(Bn * 32 * 32) / 256, 256, 0, stream>>>(qbuf, kbuf, vbuf, x, gamma, out);
}
